// ProcessorStep_48447231099239
// MI455X (gfx1250) — matrix operand order repaired and checked by reading, not compiled, not run on hardware
//
#include <hip/hip_runtime.h>
#include <hip/hip_bf16.h>

// ---------------------------------------------------------------------------
// GNN processor step on gfx1250 (MI455X): bf16 WMMA MLPs, fused gather +
// LayerNorm + atomic mean-aggregation. MT=64 rows/block to halve L2 weight
// traffic; each wave: 2 N-tiles x 4 row-tiles sharing every B fragment.
// ---------------------------------------------------------------------------

typedef __attribute__((ext_vector_type(16))) __bf16 v16bf;
typedef __attribute__((ext_vector_type(8)))  float  v8f;

union AF { v16bf v; unsigned int d[8]; };
union BF { v16bf v; uint4 q[2]; };

#define HD    256
#define NDIM  128
#define MT    64         // rows (edges or nodes) per block (4 row-tiles)
#define HSTR  264        // LDS stride for 256-wide hidden tiles (bf16 elems)
#define OSTR  132        // LDS stride for 128-wide f32 output tile
#define EXSTR 392        // 384 + 8 pad (bf16 elems)

__device__ __forceinline__ unsigned short f2bf(float f) {
    unsigned int u = __float_as_uint(f);
    unsigned int r = u + 0x7FFFu + ((u >> 16) & 1u);   // round-to-nearest-even
    return (unsigned short)(r >> 16);
}

__device__ __forceinline__ v8f wmma_bf16(const AF& a, const BF& b, v8f c) {
    return __builtin_amdgcn_wmma_f32_16x16x32_bf16(
        /*neg_a=*/false, a.v, /*neg_b=*/false, b.v,
        /*c_mod=*/(short)0, c, /*reuse_a=*/false, /*reuse_b=*/false);
}

// A fragment (16x32 bf16) from LDS, row-major with stride `xstr` (elements).
__device__ __forceinline__ void load_a(AF& f, const unsigned short* sA,
                                       int rowBase, int xstr, int kbase, int lane) {
    const int m    = rowBase + (lane & 15);
    const int half = lane >> 4;
    const unsigned short* p = sA + m * xstr + kbase;
#pragma unroll
    for (int j = 0; j < 8; ++j) {
        const int koff = 16 * half + 2 * j;
        f.d[j] = *(const unsigned int*)(p + koff);
    }
}

// B fragment (32x16 bf16) from pre-transposed weights Wt[n*K + k] (bf16).
__device__ __forceinline__ void load_b(BF& f, const unsigned short* Wt,
                                       int K, int ntile, int kbase, int lane) {
    const int n    = lane & 15;
    const int half = lane >> 4;
    const uint4* p = (const uint4*)(Wt + (size_t)(ntile * 16 + n) * K + kbase + 16 * half);
    f.q[0] = p[0];
    f.q[1] = p[1];
}

__device__ __forceinline__ v8f bias_acc(const float* b, int ntile, int lane) {
    const float bv = b[ntile * 16 + (lane & 15)];
    v8f a;
#pragma unroll
    for (int v = 0; v < 8; ++v) a[v] = bv;
    return a;
}

__device__ __forceinline__ void store_relu_bf16(const v8f& acc, unsigned short* sH,
                                                int rowBase, int hstr, int ntile, int lane) {
    const int n    = ntile * 16 + (lane & 15);
    const int half = lane >> 4;
#pragma unroll
    for (int v = 0; v < 8; ++v) {
        sH[(rowBase + half * 8 + v) * hstr + n] = f2bf(fmaxf(acc[v], 0.f));
    }
}

__device__ __forceinline__ void store_f32(const v8f& acc, float* sO,
                                          int rowBase, int ostr, int ntile, int lane) {
    const int n    = ntile * 16 + (lane & 15);
    const int half = lane >> 4;
#pragma unroll
    for (int v = 0; v < 8; ++v) {
        sO[(rowBase + half * 8 + v) * ostr + n] = acc[v];
    }
}

// ---------------------------------------------------------------------------
// 3-layer MLP on a 64-row tile: [64 x K1] -> relu 256 -> relu 256 -> 128 (sO).
// Layers 1/2: wave w owns N-tiles {w, w+8}; 8 accumulators (2 nt x 4 rt).
// Layer 3:   wave w owns N-tile w;          4 accumulators.
// Every B fragment feeds 4 row-tiles; every A fragment feeds 2 N-tiles.
// ---------------------------------------------------------------------------
template <int K1>
__device__ void mlp3_tile(const unsigned short* sX, int xstr,
                          unsigned short* sH1, unsigned short* sH2, float* sO,
                          const unsigned short* W1t, const float* b1,
                          const unsigned short* W2t, const float* b2,
                          const unsigned short* W3t, const float* b3) {
    const int lane = threadIdx.x & 31;
    const int wave = threadIdx.x >> 5;

    // ---- Layer 1: X[64xK1] @ W1[K1x256] ----
    {
        v8f acc[2][4];
#pragma unroll
        for (int ti = 0; ti < 2; ++ti) {
            v8f b = bias_acc(b1, wave + 8 * ti, lane);
#pragma unroll
            for (int r = 0; r < 4; ++r) acc[ti][r] = b;
        }
        for (int kb = 0; kb < K1 / 32; ++kb) {
            AF fa[4];
#pragma unroll
            for (int r = 0; r < 4; ++r) load_a(fa[r], sX, 16 * r, xstr, kb * 32, lane);
#pragma unroll
            for (int ti = 0; ti < 2; ++ti) {
                BF bf; load_b(bf, W1t, K1, wave + 8 * ti, kb * 32, lane);
#pragma unroll
                for (int r = 0; r < 4; ++r) acc[ti][r] = wmma_bf16(fa[r], bf, acc[ti][r]);
            }
        }
#pragma unroll
        for (int ti = 0; ti < 2; ++ti)
#pragma unroll
            for (int r = 0; r < 4; ++r)
                store_relu_bf16(acc[ti][r], sH1, 16 * r, HSTR, wave + 8 * ti, lane);
    }
    __syncthreads();

    // ---- Layer 2: H1[64x256] @ W2[256x256] ----
    {
        v8f acc[2][4];
#pragma unroll
        for (int ti = 0; ti < 2; ++ti) {
            v8f b = bias_acc(b2, wave + 8 * ti, lane);
#pragma unroll
            for (int r = 0; r < 4; ++r) acc[ti][r] = b;
        }
        for (int kb = 0; kb < 8; ++kb) {
            AF fa[4];
#pragma unroll
            for (int r = 0; r < 4; ++r) load_a(fa[r], sH1, 16 * r, HSTR, kb * 32, lane);
#pragma unroll
            for (int ti = 0; ti < 2; ++ti) {
                BF bf; load_b(bf, W2t, 256, wave + 8 * ti, kb * 32, lane);
#pragma unroll
                for (int r = 0; r < 4; ++r) acc[ti][r] = wmma_bf16(fa[r], bf, acc[ti][r]);
            }
        }
#pragma unroll
        for (int ti = 0; ti < 2; ++ti)
#pragma unroll
            for (int r = 0; r < 4; ++r)
                store_relu_bf16(acc[ti][r], sH2, 16 * r, HSTR, wave + 8 * ti, lane);
    }
    __syncthreads();

    // ---- Layer 3: H2[64x256] @ W3[256x128] -> f32 (overlays sX; dead now) ----
    {
        const int t = wave;
        v8f acc[4];
        {
            v8f b = bias_acc(b3, t, lane);
#pragma unroll
            for (int r = 0; r < 4; ++r) acc[r] = b;
        }
        for (int kb = 0; kb < 8; ++kb) {
            BF bf; load_b(bf, W3t, 256, t, kb * 32, lane);
#pragma unroll
            for (int r = 0; r < 4; ++r) {
                AF fa; load_a(fa, sH2, 16 * r, HSTR, kb * 32, lane);
                acc[r] = wmma_bf16(fa, bf, acc[r]);
            }
        }
#pragma unroll
        for (int r = 0; r < 4; ++r) store_f32(acc[r], sO, 16 * r, OSTR, t, lane);
    }
    __syncthreads();
}

__device__ __forceinline__ float wave_sum32(float s) {
#pragma unroll
    for (int m = 16; m >= 1; m >>= 1) s += __shfl_xor(s, m, 32);
    return s;
}

// ---------------------------------------------------------------------------
// Utility kernels
// ---------------------------------------------------------------------------
__global__ __launch_bounds__(256) void zero_ws_kernel(float* p, size_t n) {
    size_t i = (size_t)blockIdx.x * blockDim.x + threadIdx.x;
    if (i < n) p[i] = 0.f;
}

// Wt[n*K + k] = bf16(W[k*N + n])
__global__ __launch_bounds__(256) void convert_wt_kernel(const float* __restrict__ W,
                                                         unsigned short* __restrict__ Wt,
                                                         int K, int Nc) {
    int o = blockIdx.x * blockDim.x + threadIdx.x;
    if (o >= K * Nc) return;
    int n = o / K, k = o % K;
    Wt[o] = f2bf(W[k * Nc + n]);
}

// ---------------------------------------------------------------------------
// Edge kernel: e' = LN(MLP(concat(h_src, h_dst, e))) + e ; atomic agg into dst
// ---------------------------------------------------------------------------
__global__ __launch_bounds__(256) void edge_kernel(
    const float* __restrict__ node_emb, const float* __restrict__ edge_emb,
    const int* __restrict__ src, const int* __restrict__ dst, int E,
    const unsigned short* __restrict__ eW1t, const float* __restrict__ eb1,
    const unsigned short* __restrict__ eW2t, const float* __restrict__ eb2,
    const unsigned short* __restrict__ eW3t, const float* __restrict__ eb3,
    const float* __restrict__ eg, const float* __restrict__ ebeta,
    float* __restrict__ out_edge, float* __restrict__ agg, float* __restrict__ cnt) {

    __shared__ __align__(16) unsigned char smem[MT * EXSTR * 2 + 2 * MT * HSTR * 2];
    unsigned short* sX  = (unsigned short*)smem;
    unsigned short* sH1 = (unsigned short*)(smem + MT * EXSTR * 2);
    unsigned short* sH2 = (unsigned short*)(smem + MT * EXSTR * 2 + MT * HSTR * 2);
    float* sO = (float*)smem;                       // overlays sX (dead after L1)

    const int e0 = blockIdx.x * MT;

    // Gather + f32->bf16: X = [h_src | h_dst | e]  (64 x 384)
    for (int idx = threadIdx.x; idx < MT * 384; idx += 256) {
        int r = idx / 384, c = idx - r * 384;
        int e = e0 + r;
        float val = 0.f;
        if (e < E) {
            if (c < 128)      val = node_emb[(size_t)src[e] * NDIM + c];
            else if (c < 256) val = node_emb[(size_t)dst[e] * NDIM + (c - 128)];
            else              val = edge_emb[(size_t)e * NDIM + (c - 256)];
        }
        sX[r * EXSTR + c] = f2bf(val);
    }
    __syncthreads();

    mlp3_tile<384>(sX, EXSTR, sH1, sH2, sO, eW1t, eb1, eW2t, eb2, eW3t, eb3);

    // LayerNorm + gamma/beta + residual + store + atomic aggregation
    const int lane = threadIdx.x & 31;
    const int wave = threadIdx.x >> 5;
    for (int r = wave * 8; r < wave * 8 + 8; ++r) {
        const int e = e0 + r;
        if (e >= E) continue;
        float v[4];
        float s = 0.f;
#pragma unroll
        for (int i = 0; i < 4; ++i) { v[i] = sO[r * OSTR + lane + 32 * i]; s += v[i]; }
        const float mu = wave_sum32(s) * (1.f / 128.f);
        float q = 0.f;
#pragma unroll
        for (int i = 0; i < 4; ++i) { float d = v[i] - mu; q += d * d; }
        const float var = wave_sum32(q) * (1.f / 128.f);
        const float inv = rsqrtf(var + 1e-5f);
        const int dn = dst[e];
#pragma unroll
        for (int i = 0; i < 4; ++i) {
            const int c = lane + 32 * i;
            const float o = (v[i] - mu) * inv * eg[c] + ebeta[c]
                          + edge_emb[(size_t)e * NDIM + c];
            out_edge[(size_t)e * NDIM + c] = o;
            atomicAdd(&agg[(size_t)dn * NDIM + c], o);
        }
        if (lane == 0) atomicAdd(&cnt[dn], 1.f);
    }
}

// ---------------------------------------------------------------------------
// Node kernel: h' = LN(MLP(concat(h, agg/max(count,1)))) + h
// ---------------------------------------------------------------------------
__global__ __launch_bounds__(256) void node_kernel(
    const float* __restrict__ node_emb, const float* __restrict__ agg,
    const float* __restrict__ cnt, int N,
    const unsigned short* __restrict__ nW1t, const float* __restrict__ nb1,
    const unsigned short* __restrict__ nW2t, const float* __restrict__ nb2,
    const unsigned short* __restrict__ nW3t, const float* __restrict__ nb3,
    const float* __restrict__ ng, const float* __restrict__ nbeta,
    float* __restrict__ out_node) {

    __shared__ __align__(16) unsigned char smem[3 * MT * HSTR * 2];
    unsigned short* sX  = (unsigned short*)smem;
    unsigned short* sH1 = (unsigned short*)(smem + MT * HSTR * 2);
    unsigned short* sH2 = (unsigned short*)(smem + 2 * MT * HSTR * 2);
    float* sO = (float*)smem;                       // overlays sX

    const int n0 = blockIdx.x * MT;

    // Build X = [h | agg/max(cnt,1)]  (64 x 256)
    for (int idx = threadIdx.x; idx < MT * 256; idx += 256) {
        int r = idx >> 8, c = idx & 255;
        int n = n0 + r;
        float val = 0.f;
        if (n < N) {
            if (c < 128) {
                val = node_emb[(size_t)n * NDIM + c];
            } else {
                const float ct = fmaxf(cnt[n], 1.f);
                val = agg[(size_t)n * NDIM + (c - 128)] / ct;
            }
        }
        sX[r * HSTR + c] = f2bf(val);
    }
    __syncthreads();

    mlp3_tile<256>(sX, HSTR, sH1, sH2, sO, nW1t, nb1, nW2t, nb2, nW3t, nb3);

    const int lane = threadIdx.x & 31;
    const int wave = threadIdx.x >> 5;
    for (int r = wave * 8; r < wave * 8 + 8; ++r) {
        const int n = n0 + r;
        if (n >= N) continue;
        float v[4];
        float s = 0.f;
#pragma unroll
        for (int i = 0; i < 4; ++i) { v[i] = sO[r * OSTR + lane + 32 * i]; s += v[i]; }
        const float mu = wave_sum32(s) * (1.f / 128.f);
        float q = 0.f;
#pragma unroll
        for (int i = 0; i < 4; ++i) { float d = v[i] - mu; q += d * d; }
        const float var = wave_sum32(q) * (1.f / 128.f);
        const float inv = rsqrtf(var + 1e-5f);
#pragma unroll
        for (int i = 0; i < 4; ++i) {
            const int c = lane + 32 * i;
            const float o = (v[i] - mu) * inv * ng[c] + nbeta[c]
                          + node_emb[(size_t)n * NDIM + c];
            out_node[(size_t)n * NDIM + c] = o;
        }
    }
}

// ---------------------------------------------------------------------------
extern "C" void kernel_launch(void* const* d_in, const int* in_sizes, int n_in,
                              void* d_out, int out_size, void* d_ws, size_t ws_size,
                              hipStream_t stream) {
    const float* node_emb = (const float*)d_in[0];
    const float* edge_emb = (const float*)d_in[1];
    const int*   src      = (const int*)d_in[2];
    const int*   dst      = (const int*)d_in[3];
    const int N = in_sizes[0] / NDIM;
    const int E = in_sizes[2];

    const float* eW1 = (const float*)d_in[5];
    const float* eb1 = (const float*)d_in[6];
    const float* eW2 = (const float*)d_in[7];
    const float* eb2 = (const float*)d_in[8];
    const float* eW3 = (const float*)d_in[9];
    const float* eb3 = (const float*)d_in[10];
    const float* eg    = (const float*)d_in[11];
    const float* ebeta = (const float*)d_in[12];
    const float* nW1 = (const float*)d_in[13];
    const float* nb1 = (const float*)d_in[14];
    const float* nW2 = (const float*)d_in[15];
    const float* nb2 = (const float*)d_in[16];
    const float* nW3 = (const float*)d_in[17];
    const float* nb3 = (const float*)d_in[18];
    const float* ng    = (const float*)d_in[19];
    const float* nbeta = (const float*)d_in[20];

    float* out_node = (float*)d_out;
    float* out_edge = out_node + (size_t)N * NDIM;

    // Workspace: agg (N*128 f32) | cnt (N f32) | bf16-transposed weights
    float* agg = (float*)d_ws;
    float* cnt = agg + (size_t)N * NDIM;
    unsigned short* wbf  = (unsigned short*)(cnt + N);
    unsigned short* eW1t = wbf;                       // 384*256
    unsigned short* eW2t = eW1t + 384 * 256;          // 256*256
    unsigned short* eW3t = eW2t + 256 * 256;          // 256*128
    unsigned short* nW1t = eW3t + 256 * 128;          // 256*256
    unsigned short* nW2t = nW1t + 256 * 256;          // 256*256
    unsigned short* nW3t = nW2t + 256 * 256;          // 256*128

    // 1) zero agg + cnt
    {
        const size_t n = (size_t)N * (NDIM + 1);
        zero_ws_kernel<<<(unsigned)((n + 255) / 256), 256, 0, stream>>>(agg, n);
    }
    // 2) bf16 convert + transpose weights (L2-resident afterwards)
    convert_wt_kernel<<<(384 * 256 + 255) / 256, 256, 0, stream>>>(eW1, eW1t, 384, 256);
    convert_wt_kernel<<<(256 * 256 + 255) / 256, 256, 0, stream>>>(eW2, eW2t, 256, 256);
    convert_wt_kernel<<<(256 * 128 + 255) / 256, 256, 0, stream>>>(eW3, eW3t, 256, 128);
    convert_wt_kernel<<<(256 * 256 + 255) / 256, 256, 0, stream>>>(nW1, nW1t, 256, 256);
    convert_wt_kernel<<<(256 * 256 + 255) / 256, 256, 0, stream>>>(nW2, nW2t, 256, 256);
    convert_wt_kernel<<<(256 * 128 + 255) / 256, 256, 0, stream>>>(nW3, nW3t, 256, 128);

    // 3) edge MLP + LN + residual + atomic aggregation
    edge_kernel<<<(E + MT - 1) / MT, 256, 0, stream>>>(
        node_emb, edge_emb, src, dst, E,
        eW1t, eb1, eW2t, eb2, eW3t, eb3, eg, ebeta,
        out_edge, agg, cnt);

    // 4) node MLP + LN + residual
    node_kernel<<<(N + MT - 1) / MT, 256, 0, stream>>>(
        node_emb, agg, cnt, N,
        nW1t, nb1, nW2t, nb2, nW3t, nb3, ng, nbeta,
        out_node);
}
